// CustomESN_8873402434092
// MI455X (gfx1250) — matrix operand order repaired and checked by reading, not compiled, not run on hardware
//
#include <hip/hip_runtime.h>
#include <math.h>

// ---------------------------------------------------------------------------
// ESN scan for MI455X (gfx1250), wave32 + WMMA bf16.
//
// state_{t+1} = 0.5*state + 0.5*tanh(state @ W_res + x_t @ W_in)
//   B=512 T=512 R=1024 I=64 O=64.  ~550 GFLOP of recurrent GEMM.
// The scan is independent per batch row: each workgroup owns 16 rows, keeps
// state resident in LDS for all 512 steps (LDS barriers only, no global
// sync), and fuses the input projection as extra K (K_total = 1088).
// ---------------------------------------------------------------------------

typedef __attribute__((ext_vector_type(16))) __bf16 v16bf;
typedef __attribute__((ext_vector_type(8)))  __bf16 v8bf;
typedef __attribute__((ext_vector_type(8)))  float  v8f;

#define ESN_B    512
#define ESN_T    512
#define ESN_I    64
#define ESN_R    1024
#define ESN_O    64
#define KTOT     1088            // R + I, fused K dimension
#define NKT      34              // KTOT / 32 (K-tiles for 16x16x32 WMMA)
#define NNT      64              // R / 16   (N-tiles of recurrent GEMM)
#define FRAG     512             // bf16 elements per 32x16 B fragment
#define WRES_FRAGS (NKT * NNT)   // 2176 fragments for [W_res; W_in]
#define WOUT_KT  32              // R / 32
#define WOUT_NT  4               // O / 16
#define WOUT_FRAGS (WOUT_KT * WOUT_NT)
#define PACK_TOT (WRES_FRAGS * FRAG + WOUT_FRAGS * FRAG)   // 1,179,648 bf16
#define SMEM_BYTES (16 * KTOT * 2 + 16 * ESN_R * 4)        // 100,352 B

// Branch-free tanh: copysign(1 - 2/(exp2(2*log2e*|x|) + 1), x).
// Raw v_exp_f32 saturates to +inf for large |x| -> rcp -> 0 -> tanh = +/-1,
// so no piecewise range handling (and no EXEC divergence) is needed.
__device__ __forceinline__ float fast_tanh(float x) {
  float ax = __builtin_fabsf(x);
  float e  = __builtin_amdgcn_exp2f(ax * 2.8853900817779268f);  // 2/ln(2)
  float t  = 1.0f - 2.0f * __builtin_amdgcn_rcpf(e + 1.0f);
  return __builtin_copysignf(t, x);
}

// ---------------------------------------------------------------------------
// Prepack [W_res; W_in] and W_out (f32, row-major [K][N]) into bf16 fragments
// in a contiguous-K 32x16 B order (this file's choice, not the ISA table order):
//   lane L holds column n = L&15; lanes 0-15 carry K=0..15, lanes 16-31 carry
//   K=16..31; within a lane, element e (0..15) is K = (L>>4)*16 + e.
// Stored as pack[frag][lane][16] so the hot loop does 2x global_load_b128.
// ---------------------------------------------------------------------------
__global__ void esn_pack_kernel(const float* __restrict__ Wres,
                                const float* __restrict__ Win,
                                const float* __restrict__ Wout,
                                __bf16* __restrict__ pack) {
  int idx = blockIdx.x * 256 + threadIdx.x;
  if (idx >= PACK_TOT) return;
  float v;
  if (idx < WRES_FRAGS * FRAG) {
    int f = idx >> 9, r = idx & 511;
    int L = r >> 4, e = r & 15;
    int kt = f / NNT, nt = f % NNT;
    int k = kt * 32 + ((L >> 4) << 4) + e;     // 0..1087
    int n = nt * 16 + (L & 15);                // 0..1023
    v = (k < ESN_R) ? Wres[(size_t)k * ESN_R + n]
                    : Win[(size_t)(k - ESN_R) * ESN_R + n];
  } else {
    int idx2 = idx - WRES_FRAGS * FRAG;
    int f = idx2 >> 9, r = idx2 & 511;
    int L = r >> 4, e = r & 15;
    int kt = f / WOUT_NT, nt = f % WOUT_NT;
    int k = kt * 32 + ((L >> 4) << 4) + e;     // 0..1023
    int n = nt * 16 + (L & 15);                // 0..63
    v = Wout[(size_t)k * ESN_O + n];
  }
  pack[idx] = (__bf16)v;
}

// A fragment (16x32 bf16) from LDS state, in contiguous-K order:
//   lane L -> row (L&15); 16 elems at K = kt*32 + (L>>4)*16 + e
//   (matches B pack).  Two 16B ds_load_b128.
__device__ __forceinline__ v16bf load_a(const __bf16* stateA, int lane, int kt) {
  const __bf16* p = stateA + (lane & 15) * KTOT + kt * 32 + ((lane >> 4) << 4);
  v8bf lo = *(const v8bf*)(p);
  v8bf hi = *(const v8bf*)(p + 8);
  return __builtin_shufflevector(lo, hi, 0, 1, 2, 3, 4, 5, 6, 7,
                                         8, 9, 10, 11, 12, 13, 14, 15);
}

// B fragment (32x16 bf16) from prepacked global: 32 contiguous bytes per lane.
__device__ __forceinline__ v16bf load_b(const __bf16* __restrict__ pack,
                                        int frag, int lane) {
  const v8bf* p = (const v8bf*)(pack + (size_t)frag * FRAG + lane * 16);
  v8bf lo = p[0];
  v8bf hi = p[1];
  return __builtin_shufflevector(lo, hi, 0, 1, 2, 3, 4, 5, 6, 7,
                                         8, 9, 10, 11, 12, 13, 14, 15);
}

// ---------------------------------------------------------------------------
// Persistent scan: 32 blocks x 1024 threads (32 wave32s = 8 waves/SIMD).
// Block b owns batch rows [16b, 16b+16). Wave w computes N-tiles {2w, 2w+1}.
// Per step: 34 K-tiles x 2 N-tiles = 68 v_wmma_f32_16x16x32_bf16 per wave.
// ---------------------------------------------------------------------------
__global__ void __launch_bounds__(1024, 1)
esn_scan_kernel(const float* __restrict__ inp,        // [B][T][I] f32
                const __bf16* __restrict__ pack,      // prepacked weights
                float* __restrict__ out) {            // [B][O] f32
  extern __shared__ char smem[];
  __bf16* stateA = (__bf16*)smem;                       // [16][KTOT] bf16 (A operand)
  float*  stateF = (float*)(smem + 16 * KTOT * 2);      // [16][R] f32 (exact state)

  const int tid  = threadIdx.x;
  const int lane = tid & 31;
  const int wave = tid >> 5;        // 0..31
  const int blk  = blockIdx.x;      // 0..31

  // Zero-init state.
  for (int i = tid; i < 16 * KTOT; i += 1024) stateA[i] = (__bf16)0.0f;
  for (int i = tid; i < 16 * ESN_R; i += 1024) stateF[i] = 0.0f;

  // Stage input slice for t = 0 into A columns [R, R+I).
  {
    int row = tid >> 6, ii = tid & 63;   // 16 rows x 64 cols, one elem/thread
    stateA[row * KTOT + ESN_R + ii] =
        (__bf16)inp[(size_t)(blk * 16 + row) * ESN_T * ESN_I + ii];
  }
  __syncthreads();

  const int nt0 = wave * 2;
  const int nt1 = nt0 + 1;
  const int ncol0 = nt0 * 16 + (lane & 15);
  const int ncol1 = nt1 * 16 + (lane & 15);
  const int rowoff = (lane >> 4) * 8;     // C-layout: lanes>=16 hold M=i+8

  for (int t = 0; t < ESN_T; ++t) {
    v8f acc0 = {};
    v8f acc1 = {};
    #pragma unroll 2
    for (int kt = 0; kt < NKT; ++kt) {
      v16bf a  = load_a(stateA, lane, kt);
      v16bf b0 = load_b(pack, kt * NNT + nt0, lane);
      v16bf b1 = load_b(pack, kt * NNT + nt1, lane);
      acc0 = __builtin_amdgcn_wmma_f32_16x16x32_bf16(
          false, a, false, b0, (short)0, acc0, false, false);
      acc1 = __builtin_amdgcn_wmma_f32_16x16x32_bf16(
          false, a, false, b1, (short)0, acc1, false, false);
    }
    __syncthreads();   // all waves done READING stateA before anyone writes

    // state' = 0.5*state + 0.5*tanh(acc); write f32 truth + bf16 A operand.
    #pragma unroll
    for (int i2 = 0; i2 < 8; ++i2) {
      int row = rowoff + i2;
      float o0 = stateF[row * ESN_R + ncol0];
      float n0 = __builtin_fmaf(0.5f, o0, 0.5f * fast_tanh(acc0[i2]));
      stateF[row * ESN_R + ncol0] = n0;
      stateA[row * KTOT + ncol0]  = (__bf16)n0;
      float o1 = stateF[row * ESN_R + ncol1];
      float n1 = __builtin_fmaf(0.5f, o1, 0.5f * fast_tanh(acc1[i2]));
      stateF[row * ESN_R + ncol1] = n1;
      stateA[row * KTOT + ncol1]  = (__bf16)n1;
    }
    // Stage next step's input slice (disjoint LDS region, same barrier).
    if (t + 1 < ESN_T) {
      int row = tid >> 6, ii = tid & 63;
      stateA[row * KTOT + ESN_R + ii] =
          (__bf16)inp[(size_t)(blk * 16 + row) * ESN_T * ESN_I +
                      (size_t)(t + 1) * ESN_I + ii];
    }
    __syncthreads();   // writes visible before next step's reads
  }

  // Readout: out[16x64] = state[16x1024] @ W_out. Waves 0..3, one N-tile each.
  if (wave < 4) {
    v8f acc = {};
    #pragma unroll 2
    for (int kt = 0; kt < WOUT_KT; ++kt) {
      v16bf a = load_a(stateA, lane, kt);
      v16bf b = load_b(pack, WRES_FRAGS + kt * WOUT_NT + wave, lane);
      acc = __builtin_amdgcn_wmma_f32_16x16x32_bf16(
          false, a, false, b, (short)0, acc, false, false);
    }
    int col = wave * 16 + (lane & 15);
    #pragma unroll
    for (int i2 = 0; i2 < 8; ++i2) {
      out[(size_t)(blk * 16 + rowoff + i2) * ESN_O + col] = acc[i2];
    }
  }
}

// ---------------------------------------------------------------------------
extern "C" void kernel_launch(void* const* d_in, const int* in_sizes, int n_in,
                              void* d_out, int out_size, void* d_ws, size_t ws_size,
                              hipStream_t stream) {
  const float* inp  = (const float*)d_in[0];   // [512][512][64]
  const float* Wres = (const float*)d_in[1];   // [1024][1024]
  const float* Win  = (const float*)d_in[2];   // [64][1024]
  const float* Wout = (const float*)d_in[3];   // [1024][64]
  float* out = (float*)d_out;                  // [512][64]
  __bf16* pack = (__bf16*)d_ws;                // ~2.36 MB of fragments

  esn_pack_kernel<<<(PACK_TOT + 255) / 256, 256, 0, stream>>>(Wres, Win, Wout, pack);
  esn_scan_kernel<<<ESN_B / 16, 1024, SMEM_BYTES, stream>>>(inp, pack, out);
}
